// RelateModel_652835029255
// MI455X (gfx1250) — compile-verified, matrix operand order checked by reading, not run on hardware
//
#include <hip/hip_runtime.h>
#include <hip/hip_bf16.h>

// ---------------------------------------------------------------------------
// RGCN (3 layers, mean aggr) + BN/ELU + residual + mean-pool + 2-layer MLP
// Strategy:
//   * transform-then-scatter:  y_r = h @ W_r (bf16 WMMA), out = h @ W_root + b
//     then per edge: out[dst] += y_{type}[src] * inv_cnt[type][dst]  (f32 atomics)
//   * all GEMMs via v_wmma_f32_16x16x32_bf16 (wave32), f32 accumulate
//   * edge-degree reciprocals computed once per launch (edge list is fixed)
// ---------------------------------------------------------------------------

typedef __attribute__((ext_vector_type(16))) __bf16 v16bf;
typedef __attribute__((ext_vector_type(8)))  float  v8f;

union Frag { uint4 q[2]; v16bf v; };

__device__ __forceinline__ unsigned short f32_to_bf16(float f) {
  unsigned int u = __float_as_uint(f);
  u = (u + 0x7FFFu + ((u >> 16) & 1u)) >> 16;   // round-to-nearest-even
  return (unsigned short)u;
}
__device__ __forceinline__ float bf16_to_f32(unsigned short s) {
  return __uint_as_float(((unsigned int)s) << 16);
}

// ----------------------------- small utility kernels -----------------------

__global__ void k_zero_f32(float* __restrict__ p, int n) {
  int i = blockIdx.x * blockDim.x + threadIdx.x;
  if (i < n) p[i] = 0.0f;
}

// f32 -> bf16 with zero-padding of padded rows
__global__ void k_cvt_bf16(const float* __restrict__ h, unsigned short* __restrict__ hb,
                           int n_valid, int n_total) {
  int i = blockIdx.x * blockDim.x + threadIdx.x;
  if (i < n_total) hb[i] = (i < n_valid) ? f32_to_bf16(h[i]) : (unsigned short)0;
}

// Transpose + convert all 15 weight matrices: wt[l][s][n][k] = W[l][s][k][n]
__global__ void k_prep_wt(const float* __restrict__ rel, const float* __restrict__ root,
                          unsigned short* __restrict__ wt) {
  int i = blockIdx.x * blockDim.x + threadIdx.x;
  if (i >= 3 * 5 * 128 * 128) return;
  int k = i & 127;
  int n = (i >> 7) & 127;
  int s = (i >> 14) % 5;
  int l = (i >> 14) / 5;
  float v = (s == 0) ? root[((size_t)l * 128 + k) * 128 + n]
                     : rel[(((size_t)l * 4 + (s - 1)) * 128 + k) * 128 + n];
  wt[i] = f32_to_bf16(v);
}

// per (relation,dst) in-degree counts
__global__ void k_count(const int* __restrict__ ei, const int* __restrict__ et,
                        float* __restrict__ cnt, int E, int N) {
  int e = blockIdx.x * blockDim.x + threadIdx.x;
  if (e >= E) return;
  atomicAdd(&cnt[(size_t)et[e] * N + ei[E + e]], 1.0f);
}
__global__ void k_invcnt(float* __restrict__ c, int n) {
  int i = blockIdx.x * blockDim.x + threadIdx.x;
  if (i < n) c[i] = 1.0f / fmaxf(c[i], 1.0f);
}

// ----------------------------- WMMA GEMM -----------------------------------
// A: [Npad,128] bf16.  Wt segment: [128(n)][128(k)] bf16 (pre-transposed).
// seg 0: outF = A @ W_root + bias (f32).  seg 1..4: Y[seg-1] = A @ W_rel (bf16).
__global__ __launch_bounds__(128)
void k_gemm(const unsigned short* __restrict__ A,
            const unsigned short* __restrict__ Wt,
            const float* __restrict__ bias,
            float* __restrict__ outF,
            unsigned short* __restrict__ Y,
            int Npad, int n_rows) {
  const int seg  = blockIdx.y;
  const int wave = threadIdx.x >> 5;
  const int lane = threadIdx.x & 31;
  const int half = lane >> 4;      // k-half selector per ISA A/B layout
  const int lr   = lane & 15;      // row within 16-tile
  const int m0   = blockIdx.x * 64 + wave * 16;

  const unsigned short* arow = A + (size_t)(m0 + lr) * 128;
  const unsigned short* wseg = Wt + (size_t)seg * 16384;

  v8f acc[8];
  v8f zero = {};
#pragma unroll
  for (int t = 0; t < 8; ++t) acc[t] = zero;

#pragma unroll
  for (int k0 = 0; k0 < 128; k0 += 32) {
    const int ka = k0 + half * 8;          // lane's first 8-elem chunk
    Frag a;
    a.q[0] = *(const uint4*)(arow + ka);
    a.q[1] = *(const uint4*)(arow + ka + 16);
#pragma unroll
    for (int t = 0; t < 8; ++t) {
      const unsigned short* brow = wseg + (size_t)(t * 16 + lr) * 128;
      Frag b;
      b.q[0] = *(const uint4*)(brow + ka);
      b.q[1] = *(const uint4*)(brow + ka + 16);
      acc[t] = __builtin_amdgcn_wmma_f32_16x16x32_bf16(
          false, a.v, false, b.v, (short)0, acc[t], false, false);
    }
  }

#pragma unroll
  for (int t = 0; t < 8; ++t) {
#pragma unroll
    for (int v = 0; v < 8; ++v) {
      int row = m0 + v + half * 8;         // C/D layout: VGPR v -> M = v (+8 for hi lanes)
      int col = t * 16 + lr;
      if (row < n_rows) {
        float val = acc[t][v];
        if (seg == 0)
          outF[(size_t)row * 128 + col] = val + bias[col];
        else
          Y[((size_t)(seg - 1) * Npad + row) * 128 + col] = f32_to_bf16(val);
      }
    }
  }
}

// ----------------------------- edge scatter --------------------------------
__global__ void k_scatter(const int* __restrict__ ei, const int* __restrict__ et,
                          const unsigned short* __restrict__ Y,
                          const float* __restrict__ inv,
                          float* __restrict__ outF, int E, int N, int Npad) {
  int gid = blockIdx.x * blockDim.x + threadIdx.x;   // E*128 = 76.8M < 2^31
  int e = gid >> 7;
  int d = gid & 127;
  if (e >= E) return;
  int src = ei[e];
  int dst = ei[E + e];
  int r   = et[e];
  float w = inv[(size_t)r * N + dst];
  float v = bf16_to_f32(Y[((size_t)r * Npad + src) * 128 + d]) * w;
  atomicAdd(&outF[(size_t)dst * 128 + d], v);
}

// ----------------------------- epilogue: residual + BN + ELU ---------------
__global__ void k_epilogue(float* __restrict__ outF, const float* __restrict__ resid,
                           const float* __restrict__ gamma, const float* __restrict__ beta,
                           const float* __restrict__ mean, const float* __restrict__ var,
                           int n_elems, int use_resid, int use_bn) {
  int i = blockIdx.x * blockDim.x + threadIdx.x;
  if (i >= n_elems) return;
  float v = outF[i];
  if (use_resid) v += resid[i];
  if (use_bn) {
    int d = i & 127;
    v = (v - mean[d]) * __frsqrt_rn(var[d] + 1e-5f) * gamma[d] + beta[d];
    v = (v > 0.0f) ? v : expm1f(v);        // ELU(alpha=1)
  }
  outF[i] = v;
}

// ----------------------------- pooling (batch is sorted) -------------------
__global__ __launch_bounds__(128)
void k_pool(const float* __restrict__ h, const int* __restrict__ batch,
            float* __restrict__ sums, float* __restrict__ cnts, int N) {
  const int CH = 256;
  int d = threadIdx.x;                      // 128 threads = feature dims
  int i0 = blockIdx.x * CH;
  if (i0 >= N) return;
  int iend = (i0 + CH < N) ? i0 + CH : N;
  int g = batch[i0];
  float s = 0.0f, c = 0.0f;
  for (int i = i0; i < iend; ++i) {
    int gi = batch[i];
    if (gi != g) {                          // run boundary -> flush
      atomicAdd(&sums[(size_t)g * 128 + d], s);
      if (d == 0) atomicAdd(&cnts[g], c);
      s = 0.0f; c = 0.0f; g = gi;
    }
    s += h[(size_t)i * 128 + d];
    c += 1.0f;
  }
  atomicAdd(&sums[(size_t)g * 128 + d], s);
  if (d == 0) atomicAdd(&cnts[g], c);
}

// ----------------------------- classifier + log_softmax --------------------
__global__ __launch_bounds__(64)
void k_classifier(const float* __restrict__ sums, const float* __restrict__ cnts,
                  const float* __restrict__ w1, const float* __restrict__ b1,
                  const float* __restrict__ w2, const float* __restrict__ b2,
                  float* __restrict__ out) {
  __shared__ float pooled[128];
  __shared__ float zz[64];
  __shared__ float lg[16];
  int g = blockIdx.x;
  int t = threadIdx.x;
  float c = fmaxf(cnts[g], 1.0f);
  pooled[t]      = sums[(size_t)g * 128 + t] / c;
  pooled[t + 64] = sums[(size_t)g * 128 + t + 64] / c;
  __syncthreads();
  float a = b1[t];
  for (int k = 0; k < 128; ++k) a += pooled[k] * w1[k * 64 + t];
  zz[t] = (a > 0.0f) ? a : expm1f(a);
  __syncthreads();
  if (t < 16) {
    float a2 = b2[t];
    for (int j = 0; j < 64; ++j) a2 += zz[j] * w2[j * 16 + t];
    lg[t] = a2;
  }
  __syncthreads();
  if (t < 16) {
    float m = -1e30f;
    for (int k = 0; k < 16; ++k) m = fmaxf(m, lg[k]);
    float s = 0.0f;
    for (int k = 0; k < 16; ++k) s += __expf(lg[k] - m);
    out[(size_t)g * 16 + t] = lg[t] - m - __logf(s);
  }
}

// ---------------------------------------------------------------------------
extern "C" void kernel_launch(void* const* d_in, const int* in_sizes, int n_in,
                              void* d_out, int out_size, void* d_ws, size_t ws_size,
                              hipStream_t stream) {
  (void)in_sizes; (void)n_in; (void)out_size; (void)ws_size;
  const int N = 100000, E = 600000, G = 256;
  const int Npad = 100032;                 // multiple of 64 for GEMM tiling

  const float* x      = (const float*)d_in[0];
  const int*   ei     = (const int*)d_in[1];   // [2,E]: src then dst
  const int*   et     = (const int*)d_in[2];
  const int*   batch  = (const int*)d_in[3];
  const float* rel_w  = (const float*)d_in[4];
  const float* root_w = (const float*)d_in[5];
  const float* bias   = (const float*)d_in[6];
  const float* gamma  = (const float*)d_in[7];
  const float* beta   = (const float*)d_in[8];
  const float* mean   = (const float*)d_in[9];
  const float* var    = (const float*)d_in[10];
  const float* w1     = (const float*)d_in[11];
  const float* b1     = (const float*)d_in[12];
  const float* w2     = (const float*)d_in[13];
  const float* b2     = (const float*)d_in[14];
  float* out = (float*)d_out;

  // workspace carve-up (~233 MB total)
  char* ws = (char*)d_ws;
  size_t off = 0;
  auto carve = [&](size_t bytes) -> void* {
    void* p = (void*)(ws + off);
    off += (bytes + 255) & ~(size_t)255;
    return p;
  };
  unsigned short* hb  = (unsigned short*)carve((size_t)Npad * 128 * 2);
  unsigned short* Y   = (unsigned short*)carve((size_t)4 * Npad * 128 * 2);
  float* hA           = (float*)carve((size_t)Npad * 128 * 4);
  float* hB           = (float*)carve((size_t)Npad * 128 * 4);
  unsigned short* Wt  = (unsigned short*)carve((size_t)3 * 5 * 16384 * 2);
  float* inv          = (float*)carve((size_t)4 * N * 4);
  float* psum         = (float*)carve((size_t)G * 128 * 4);
  float* pcnt         = (float*)carve((size_t)G * 4);

  // one-time (per launch) prep
  k_prep_wt<<<(3 * 5 * 16384 + 255) / 256, 256, 0, stream>>>(rel_w, root_w, Wt);
  k_zero_f32<<<(4 * N + 255) / 256, 256, 0, stream>>>(inv, 4 * N);
  k_count<<<(E + 255) / 256, 256, 0, stream>>>(ei, et, inv, E, N);
  k_invcnt<<<(4 * N + 255) / 256, 256, 0, stream>>>(inv, 4 * N);

  for (int l = 0; l < 3; ++l) {
    const float* hin = (l == 0) ? x : ((l == 1) ? hA : hB);
    float* outF      = (l == 0) ? hA : ((l == 1) ? hB : hA);
    const float* resid = (l == 1) ? hA : hB;   // unused for l==0

    k_cvt_bf16<<<((size_t)Npad * 128 + 255) / 256, 256, 0, stream>>>(
        hin, hb, N * 128, Npad * 128);
    k_gemm<<<dim3(Npad / 64, 5), 128, 0, stream>>>(
        hb, Wt + (size_t)l * 5 * 16384, bias + l * 128, outF, Y, Npad, N);
    k_scatter<<<(E * 128) / 256, 256, 0, stream>>>(ei, et, Y, inv, outF, E, N, Npad);
    int bnidx = (l < 2) ? l : 0;               // guarded by use_bn
    k_epilogue<<<(N * 128 + 255) / 256, 256, 0, stream>>>(
        outF, resid, gamma + bnidx * 128, beta + bnidx * 128,
        mean + bnidx * 128, var + bnidx * 128, N * 128, l > 0, l < 2);
  }

  // pooling + classifier (final h lives in hA)
  k_zero_f32<<<(G * 128 + 255) / 256, 256, 0, stream>>>(psum, G * 128);
  k_zero_f32<<<(G + 255) / 256, 256, 0, stream>>>(pcnt, G);
  k_pool<<<(N + 255) / 256, 128, 0, stream>>>(hA, batch, psum, pcnt, N);
  k_classifier<<<G, 64, 0, stream>>>(psum, pcnt, w1, b1, w2, b2, out);
}
